// FusedMOE_26379689132707
// MI455X (gfx1250) — compile-verified, matrix operand order checked by reading, not run on hardware
//
#include <hip/hip_runtime.h>
#include <hip/hip_bf16.h>

// ---------------------------------------------------------------------------
// Problem constants (from the reference)
// ---------------------------------------------------------------------------
#define T_TOK 1024    // B*S tokens
#define HDIM  2048
#define IDIM  1408
#define EDIM  64
#define FDIM  8
#define KDIM  8
#define RDIM  8
#define TOPKN 6
#define ISHD  2816
#define KR    64      // K*R, inner dim of the low-rank mix GEMM

typedef __bf16 bf16_t;
typedef __attribute__((ext_vector_type(16))) __bf16 v16bf;
typedef __attribute__((ext_vector_type(2)))  __bf16 v2bf;
typedef __attribute__((ext_vector_type(8)))  float  v8f;

// ---------------------------------------------------------------------------
// fp32 -> bf16 (round to nearest even), and 2-element pack.
// Prefer the packed hardware convert when the builtin exists.
// ---------------------------------------------------------------------------
__device__ __forceinline__ unsigned short f2bf(float f) {
    unsigned int u = __float_as_uint(f);
    unsigned int rnd = 0x7FFFu + ((u >> 16) & 1u);
    return (unsigned short)((u + rnd) >> 16);
}

#if __has_builtin(__builtin_amdgcn_cvt_pk_bf16_f32)
__device__ __forceinline__ unsigned int pack2bf(float lo, float hi) {
    union { v2bf v; unsigned int u; } c;
    c.v = __builtin_amdgcn_cvt_pk_bf16_f32(lo, hi);
    return c.u;
}
#else
__device__ __forceinline__ unsigned int pack2bf(float lo, float hi) {
    return (unsigned int)f2bf(lo) | ((unsigned int)f2bf(hi) << 16);
}
#endif

// ---------------------------------------------------------------------------
// Generic WMMA GEMM:  C[M,N] (+)= (colScale?colScale[n]:1) * sum_k A[m,k]*B[n,k]
// A: M x Kd fp32 row-major (lda), B: N x Kd fp32 row-major (ldb)  -> C = A @ B^T
// fp32 -> bf16 conversion on the fly; v_wmma_f32_16x16x32_bf16 compute.
// Block tile 128x128, 256 threads = 8 wave32 waves (2 x 4 wave grid),
// each wave owns a 64x32 sub-tile = 4x2 WMMA accumulators.
// Double-buffered LDS; global loads for slab k+1 are in flight while the
// WMMAs for slab k execute, then converted/stored to the alternate buffer.
// Kd must be a multiple of 32 (true for all GEMMs here: 64/1408/2048/2816).
// ---------------------------------------------------------------------------
#define BM 128
#define BN 128
#define BK 32
#define LDSS 36   // shorts per LDS row: 72 B rows -> 8B-aligned b64 stores, bank-spread

__device__ __forceinline__ v16bf load_frag(const unsigned short* lds, int rbase, int lane) {
    // ISA 7.12.2 16-bit A/B fragment: lanes 0-15 row M=lane, K in {0..7,16..23};
    // lanes 16-31 same rows, K in {8..15,24..31}; 2 bf16 per VGPR.
    union { unsigned int u[8]; v16bf v; } fr;
    const int rc    = rbase + (lane & 15);
    const int khalf = (lane >> 4) << 3;
    const unsigned short* p = lds + rc * LDSS + khalf;
#pragma unroll
    for (int j = 0; j < 8; ++j) {
        int k = ((j >> 2) << 4) + ((j & 3) << 1);   // 0,2,4,6,16,18,20,22
        fr.u[j] = *reinterpret_cast<const unsigned int*>(p + k);
    }
    return fr.v;
}

__global__ __launch_bounds__(256)
void gemm_bt_wmma(const float* __restrict__ A, int lda,
                  const float* __restrict__ B, int ldb,
                  float* __restrict__ C, int ldc,
                  int M, int N, int Kd,
                  const float* __restrict__ colScale, int accumulate)
{
    __shared__ unsigned short As[2][BM * LDSS];
    __shared__ unsigned short Bs[2][BN * LDSS];

    const int tid  = threadIdx.x;
    const int lane = tid & 31;
    const int wid  = tid >> 5;
    const int wm   = wid & 1;      // 2 wave rows   -> 64 M-rows each
    const int wn   = wid >> 1;     // 4 wave cols   -> 32 N-cols each
    const int blockM = blockIdx.y * BM;
    const int blockN = blockIdx.x * BN;

    // Static per-thread staging coords: tile = 128 rows x 8 float4; 4 per thread.
    int srow[4], sc4[4];
#pragma unroll
    for (int i = 0; i < 4; ++i) {
        int idx = tid + (i << 8);
        srow[i] = idx >> 3;
        sc4[i]  = idx & 7;
    }

    float4 ra[4], rb[4];
    const float4 z4 = make_float4(0.f, 0.f, 0.f, 0.f);

    auto load_regs = [&](int k0) {
#pragma unroll
        for (int i = 0; i < 4; ++i) {
            int gr = blockM + srow[i];
            ra[i] = (gr < M)
                ? *reinterpret_cast<const float4*>(A + (size_t)gr * lda + k0 + (sc4[i] << 2))
                : z4;
            int gb = blockN + srow[i];
            rb[i] = (gb < N)
                ? *reinterpret_cast<const float4*>(B + (size_t)gb * ldb + k0 + (sc4[i] << 2))
                : z4;
        }
    };
    auto store_lds = [&](int buf) {
#pragma unroll
        for (int i = 0; i < 4; ++i) {
            uint2 pa = make_uint2(pack2bf(ra[i].x, ra[i].y), pack2bf(ra[i].z, ra[i].w));
            *reinterpret_cast<uint2*>(As[buf] + srow[i] * LDSS + (sc4[i] << 2)) = pa;
            uint2 pb = make_uint2(pack2bf(rb[i].x, rb[i].y), pack2bf(rb[i].z, rb[i].w));
            *reinterpret_cast<uint2*>(Bs[buf] + srow[i] * LDSS + (sc4[i] << 2)) = pb;
        }
    };

    v8f acc[4][2];
    v8f vzero = {};
#pragma unroll
    for (int mi = 0; mi < 4; ++mi)
#pragma unroll
        for (int ni = 0; ni < 2; ++ni)
            acc[mi][ni] = vzero;

    // Prologue: stage slab 0 into buffer 0.
    load_regs(0);
    store_lds(0);
    __syncthreads();

    int buf = 0;
    for (int k0 = 0; k0 < Kd; k0 += BK) {
        const bool has_next = (k0 + BK) < Kd;
        if (has_next) load_regs(k0 + BK);   // global loads in flight during WMMAs

        v16bf af[4], bfr[2];
#pragma unroll
        for (int mi = 0; mi < 4; ++mi)
            af[mi] = load_frag(As[buf], wm * 64 + mi * 16, lane);
#pragma unroll
        for (int ni = 0; ni < 2; ++ni)
            bfr[ni] = load_frag(Bs[buf], wn * 32 + ni * 16, lane);

#pragma unroll
        for (int mi = 0; mi < 4; ++mi)
#pragma unroll
            for (int ni = 0; ni < 2; ++ni)
                acc[mi][ni] = __builtin_amdgcn_wmma_f32_16x16x32_bf16(
                    false, af[mi], false, bfr[ni], (short)0, acc[mi][ni], false, false);

        if (has_next) store_lds(buf ^ 1);   // wait loads, convert, ds_store_b64
        __syncthreads();
        buf ^= 1;
    }

    // --- epilogue: C/D layout VGPR v -> M = v + 8*(lane>=16), N = lane&15 ---
#pragma unroll
    for (int mi = 0; mi < 4; ++mi) {
#pragma unroll
        for (int ni = 0; ni < 2; ++ni) {
            int rbase = blockM + wm * 64 + mi * 16 + ((lane >> 4) << 3);
            int col   = blockN + wn * 32 + ni * 16 + (lane & 15);
            if (col < N) {
                float cs = colScale ? colScale[col] : 1.f;
#pragma unroll
                for (int v = 0; v < 8; ++v) {
                    int row = rbase + v;
                    if (row < M) {
                        float val = acc[mi][ni][v] * cs;
                        float* p = C + (size_t)row * ldc + col;
                        *p = accumulate ? (*p + val) : val;
                    }
                }
            }
        }
    }
}

// ---------------------------------------------------------------------------
// Router: scores = softmax(x @ gwT), top-6, per-group scalar + masked softmax.
// One block (64 threads) per token; thread e owns expert e.
// ---------------------------------------------------------------------------
__global__ __launch_bounds__(64)
void router_kernel(const float* __restrict__ x, const float* __restrict__ gw,
                   const int* __restrict__ inv, float* __restrict__ soft,
                   float* __restrict__ scal)
{
    int t = blockIdx.x;
    int e = threadIdx.x;
    const float4* xt = reinterpret_cast<const float4*>(x + (size_t)t * HDIM);
    const float4* we = reinterpret_cast<const float4*>(gw + (size_t)e * HDIM);
    float s = 0.f;
    for (int i = 0; i < HDIM / 4; ++i) {
        float4 a = xt[i], b = we[i];
        s += a.x * b.x + a.y * b.y + a.z * b.z + a.w * b.w;
    }
    __shared__ float sc[EDIM];
    sc[e] = s; __syncthreads();
    float mx = -1e30f;
    for (int i = 0; i < EDIM; ++i) mx = fmaxf(mx, sc[i]);
    float ex = __expf(s - mx);
    __shared__ float pr[EDIM];
    pr[e] = ex; __syncthreads();
    float sum = 0.f;
    for (int i = 0; i < EDIM; ++i) sum += pr[i];
    float prob = ex / sum;
    __syncthreads();
    pr[e] = prob; __syncthreads();
    int rank = 0;
    for (int i = 0; i < EDIM; ++i) {
        float pi = pr[i];
        if (pi > prob || (pi == prob && i < e)) rank++;
    }
    __shared__ float wd[EDIM];
    wd[e] = (rank < TOPKN) ? prob : 0.f;
    __syncthreads();
    __shared__ float fl[EDIM];
    fl[e] = wd[inv[e]];           // flat[f,k] = w_dense[inv_mapping[f,k]]
    __syncthreads();
    int f = e >> 3, kk = e & 7;
    float scl = 0.f, m2 = -1e30f;
    for (int k = 0; k < KDIM; ++k) {
        float v = fl[f * KDIM + k];
        scl += v;
        float lv = (v == 0.f) ? -1e9f : v;
        m2 = fmaxf(m2, lv);
    }
    float ssum = 0.f;
    for (int k = 0; k < KDIM; ++k) {
        float v = fl[f * KDIM + k];
        float lv = (v == 0.f) ? -1e9f : v;
        ssum += __expf(lv - m2);
    }
    float myv = fl[e];
    float mylv = (myv == 0.f) ? -1e9f : myv;
    soft[(size_t)t * FDIM * KDIM + e] = __expf(mylv - m2) / ssum;
    if (kk == 0) scal[t * FDIM + f] = scl;
}

// ---------------------------------------------------------------------------
// Small elementwise kernels
// ---------------------------------------------------------------------------
__global__ void zero_kernel(float* p, int n) {
    int i = blockIdx.x * 256 + threadIdx.x;
    if (i < n) p[i] = 0.f;
}

// xe[t,h] = x[t,h] + sum_k soft[t,f,k] * muW_f[h,k]
__global__ void xe_kernel(const float* __restrict__ x, const float* __restrict__ soft,
                          const float* __restrict__ muW, float* __restrict__ xe, int f)
{
    int idx = blockIdx.x * 256 + threadIdx.x;
    if (idx >= T_TOK * HDIM) return;
    int t = idx >> 11, h = idx & (HDIM - 1);
    const float* w  = soft + ((size_t)t * FDIM + f) * KDIM;
    const float* mu = muW + (size_t)h * KDIM;
    float s = x[idx];
#pragma unroll
    for (int k = 0; k < KDIM; ++k) s += w[k] * mu[k];
    xe[idx] = s;
}

// aw[t,k*R+r] = soft[t,f,k] * a[t,k*R+r]
__global__ void aw_kernel(const float* __restrict__ a, const float* __restrict__ soft,
                          float* __restrict__ aw, int f)
{
    int idx = blockIdx.x * 256 + threadIdx.x;
    if (idx >= T_TOK * KR) return;
    int t = idx >> 6, k = (idx & 63) >> 3;
    aw[idx] = a[idx] * soft[((size_t)t * FDIM + f) * KDIM + k];
}

// qb (K,I2,R) -> out (I2, K*R)   so mix GEMM is aw(T x 64) @ out^T
__global__ void repack_qb_kernel(const float* __restrict__ qb, float* __restrict__ out, int I2)
{
    int idx = blockIdx.x * 256 + threadIdx.x;
    if (idx >= I2 * KR) return;
    int i = idx >> 6, kr = idx & 63, k = kr >> 3, r = kr & 7;
    out[idx] = qb[((size_t)k * I2 + i) * RDIM + r];
}

// o = silu(g) * u
__global__ void silu_mul_kernel(const float* __restrict__ g, const float* __restrict__ u,
                                float* __restrict__ o, int n)
{
    int i = blockIdx.x * 256 + threadIdx.x;
    if (i >= n) return;
    float gv = g[i];
    o[i] = (gv / (1.f + __expf(-gv))) * u[i];
}

// y[t,h] += scal[t,f] * d[t,h]
__global__ void scaled_accum_kernel(float* __restrict__ y, const float* __restrict__ d,
                                    const float* __restrict__ scal, int f)
{
    int idx = blockIdx.x * 256 + threadIdx.x;
    if (idx >= T_TOK * HDIM) return;
    int t = idx >> 11;
    y[idx] += scal[t * FDIM + f] * d[idx];
}

__global__ void add_kernel(float* __restrict__ o, const float* __restrict__ a,
                           const float* __restrict__ b, int n)
{
    int i = blockIdx.x * 256 + threadIdx.x;
    if (i < n) o[i] = a[i] + b[i];
}

// ---------------------------------------------------------------------------
// Host launcher
// ---------------------------------------------------------------------------
static inline void launch_gemm(const float* A, int lda, const float* B, int ldb,
                               float* C, int ldc, int M, int N, int Kd,
                               const float* cs, int accum, hipStream_t stream)
{
    dim3 g((N + BN - 1) / BN, (M + BM - 1) / BM);
    gemm_bt_wmma<<<g, dim3(256), 0, stream>>>(A, lda, B, ldb, C, ldc, M, N, Kd, cs, accum);
}

extern "C" void kernel_launch(void* const* d_in, const int* in_sizes, int n_in,
                              void* d_out, int out_size, void* d_ws, size_t ws_size,
                              hipStream_t stream)
{
    (void)in_sizes; (void)n_in; (void)out_size; (void)ws_size;

    const float* x        = (const float*)d_in[0];   // (T, H)
    const float* gate_w   = (const float*)d_in[1];   // (E, H)
    const int*   inv      = (const int*)  d_in[2];   // (E,)
    const float* mask_up  = (const float*)d_in[3];   // (F, H, K)
    const float* gate_W   = (const float*)d_in[4];   // (F, I, H)
    const float* gate_qa  = (const float*)d_in[5];   // (F, K, R, H)
    const float* gate_qb  = (const float*)d_in[6];   // (F, K, I, R)
    const float* gate_sc  = (const float*)d_in[7];   // (F, I)
    const float* up_W     = (const float*)d_in[8];
    const float* up_qa    = (const float*)d_in[9];
    const float* up_qb    = (const float*)d_in[10];
    const float* up_sc    = (const float*)d_in[11];
    const float* down_W   = (const float*)d_in[12];  // (F, H, I)
    const float* down_qa  = (const float*)d_in[13];  // (F, K, R, I)
    const float* down_qb  = (const float*)d_in[14];  // (F, K, H, R)
    const float* down_sc  = (const float*)d_in[15];  // (F, H)
    const float* sh_gate  = (const float*)d_in[16];  // (ISH, H)
    const float* sh_up    = (const float*)d_in[17];  // (ISH, H)
    const float* sh_down  = (const float*)d_in[18];  // (H, ISH)
    float*       out      = (float*)d_out;           // (T, H)

    // ---- workspace layout (floats). Expert phase and shared phase overlay. ----
    float* ws    = (float*)d_ws;
    float* soft  = ws;                               // T*F*K
    float* scal  = soft  + (size_t)T_TOK * FDIM * KDIM;  // T*F
    float* ybuf  = scal  + (size_t)T_TOK * FDIM;     // T*H
    float* abuf  = ybuf  + (size_t)T_TOK * HDIM;     // T*64
    float* awbuf = abuf  + (size_t)T_TOK * KR;       // T*64
    float* qbB   = awbuf + (size_t)T_TOK * KR;       // max(H,I)*64
    float* big   = qbB   + (size_t)HDIM * KR;
    // expert phase overlay
    float* xe      = big;                            // T*H
    float* gatebuf = xe      + (size_t)T_TOK * HDIM; // T*I
    float* upbuf   = gatebuf + (size_t)T_TOK * IDIM; // T*I
    float* gubuf   = upbuf   + (size_t)T_TOK * IDIM; // T*I
    float* downbuf = gubuf   + (size_t)T_TOK * IDIM; // T*H
    // shared-expert phase overlay (re-uses `big`)
    float* sg    = big;                              // T*ISH
    float* su    = sg    + (size_t)T_TOK * ISHD;     // T*ISH
    float* sprod = su    + (size_t)T_TOK * ISHD;     // T*ISH
    float* shout = sprod + (size_t)T_TOK * ISHD;     // T*H

    const int nTH  = T_TOK * HDIM;
    const int nTI  = T_TOK * IDIM;
    const int nTKR = T_TOK * KR;
    const int nTS  = T_TOK * ISHD;
    auto blk = [](int n) { return (n + 255) / 256; };

    // ---- router ----
    router_kernel<<<dim3(T_TOK), dim3(64), 0, stream>>>(x, gate_w, inv, soft, scal);
    zero_kernel<<<blk(nTH), 256, 0, stream>>>(ybuf, nTH);

    // ---- expert groups (sequential chain per f) ----
    for (int f = 0; f < FDIM; ++f) {
        const float* muW_f = mask_up + (size_t)f * HDIM * KDIM;
        const float* gW_f  = gate_W  + (size_t)f * IDIM * HDIM;
        const float* gqa_f = gate_qa + (size_t)f * KR * HDIM;   // (64, H)
        const float* gqb_f = gate_qb + (size_t)f * KDIM * IDIM * RDIM;
        const float* gs_f  = gate_sc + (size_t)f * IDIM;
        const float* uW_f  = up_W    + (size_t)f * IDIM * HDIM;
        const float* uqa_f = up_qa   + (size_t)f * KR * HDIM;
        const float* uqb_f = up_qb   + (size_t)f * KDIM * IDIM * RDIM;
        const float* us_f  = up_sc   + (size_t)f * IDIM;
        const float* dW_f  = down_W  + (size_t)f * HDIM * IDIM;
        const float* dqa_f = down_qa + (size_t)f * KR * IDIM;   // (64, I)
        const float* dqb_f = down_qb + (size_t)f * KDIM * HDIM * RDIM;
        const float* ds_f  = down_sc + (size_t)f * HDIM;

        // xe = x + soft_f @ muW^T
        xe_kernel<<<blk(nTH), 256, 0, stream>>>(x, soft, muW_f, xe, f);

        // gate = xe @ gW^T  (+ scale * low-rank mix)
        launch_gemm(xe, HDIM, gW_f, HDIM, gatebuf, IDIM, T_TOK, IDIM, HDIM, nullptr, 0, stream);
        launch_gemm(xe, HDIM, gqa_f, HDIM, abuf, KR, T_TOK, KR, HDIM, nullptr, 0, stream);
        aw_kernel<<<blk(nTKR), 256, 0, stream>>>(abuf, soft, awbuf, f);
        repack_qb_kernel<<<blk(IDIM * KR), 256, 0, stream>>>(gqb_f, qbB, IDIM);
        launch_gemm(awbuf, KR, qbB, KR, gatebuf, IDIM, T_TOK, IDIM, KR, gs_f, 1, stream);

        // up = xe @ uW^T  (+ scale * mix)
        launch_gemm(xe, HDIM, uW_f, HDIM, upbuf, IDIM, T_TOK, IDIM, HDIM, nullptr, 0, stream);
        launch_gemm(xe, HDIM, uqa_f, HDIM, abuf, KR, T_TOK, KR, HDIM, nullptr, 0, stream);
        aw_kernel<<<blk(nTKR), 256, 0, stream>>>(abuf, soft, awbuf, f);
        repack_qb_kernel<<<blk(IDIM * KR), 256, 0, stream>>>(uqb_f, qbB, IDIM);
        launch_gemm(awbuf, KR, qbB, KR, upbuf, IDIM, T_TOK, IDIM, KR, us_f, 1, stream);

        // gu = silu(gate) * up
        silu_mul_kernel<<<blk(nTI), 256, 0, stream>>>(gatebuf, upbuf, gubuf, nTI);

        // down = gu @ dW^T (+ scale * mix)
        launch_gemm(gubuf, IDIM, dW_f, IDIM, downbuf, HDIM, T_TOK, HDIM, IDIM, nullptr, 0, stream);
        launch_gemm(gubuf, IDIM, dqa_f, IDIM, abuf, KR, T_TOK, KR, IDIM, nullptr, 0, stream);
        aw_kernel<<<blk(nTKR), 256, 0, stream>>>(abuf, soft, awbuf, f);
        repack_qb_kernel<<<blk(HDIM * KR), 256, 0, stream>>>(dqb_f, qbB, HDIM);
        launch_gemm(awbuf, KR, qbB, KR, downbuf, HDIM, T_TOK, HDIM, KR, ds_f, 1, stream);

        // y += scalar_f * down
        scaled_accum_kernel<<<blk(nTH), 256, 0, stream>>>(ybuf, downbuf, scal, f);
    }

    // ---- shared expert: silu(x@gW^T) * (x@uW^T) @ dW^T ----
    launch_gemm(x, HDIM, sh_gate, HDIM, sg, ISHD, T_TOK, ISHD, HDIM, nullptr, 0, stream);
    launch_gemm(x, HDIM, sh_up,   HDIM, su, ISHD, T_TOK, ISHD, HDIM, nullptr, 0, stream);
    silu_mul_kernel<<<blk(nTS), 256, 0, stream>>>(sg, su, sprod, nTS);
    launch_gemm(sprod, ISHD, sh_down, ISHD, shout, HDIM, T_TOK, HDIM, ISHD, nullptr, 0, stream);

    // ---- out = y + sh ----
    add_kernel<<<blk(nTH), 256, 0, stream>>>(out, ybuf, shout, nTH);
}
